// SlidingWindowAttention_15212774162456
// MI455X (gfx1250) — matrix operand order repaired and checked by reading, not compiled, not run on hardware
//
#include <hip/hip_runtime.h>

typedef __attribute__((ext_vector_type(16))) _Float16 v16h;
typedef __attribute__((ext_vector_type(8)))  float    v8f;

union AFragU { v16h v; uint4 u[2]; };

__device__ __forceinline__ v8f wmma_f16(v16h a, v16h b, v8f c) {
  // D = A(16x32 f16) * B(32x16 f16) + C(16x16 f32)
  return __builtin_amdgcn_wmma_f32_16x16x32_f16(false, a, false, b, (short)0, c,
                                                false, false);
}

// gfx1250 async copy: global -> LDS, 16 bytes per lane, tracked by ASYNCcnt.
__device__ __forceinline__ void async_ld16(unsigned lds_off, const _Float16* g) {
  asm volatile("global_load_async_to_lds_b128 %0, %1, off"
               :: "v"(lds_off), "v"(g) : "memory");
}

// ---------------------------------------------------------------------------
// K1: fp32 -> f16 casts; weights transposed to [N][K] so B fragments are
// contiguous 16-byte loads.
// ---------------------------------------------------------------------------
__global__ void cast_pack_kernel(const float* __restrict__ x,
                                 const float* __restrict__ wa,
                                 const float* __restrict__ wp,
                                 _Float16* __restrict__ xh,
                                 _Float16* __restrict__ wahT,
                                 _Float16* __restrict__ wpjT,
                                 int NX, int NA, int NP, int C, int N3) {
  int total = NX + NA + NP;
  for (int idx = blockIdx.x * blockDim.x + threadIdx.x; idx < total;
       idx += gridDim.x * blockDim.x) {
    if (idx < NX) {
      xh[idx] = (_Float16)x[idx];
    } else if (idx < NX + NA) {
      int t = idx - NX;
      int kk = t / N3, n = t - kk * N3;
      wahT[(size_t)n * C + kk] = (_Float16)wa[t];
    } else {
      int t = idx - NX - NA;
      int kk = t / C, n = t - kk * C;
      wpjT[(size_t)n * C + kk] = (_Float16)wp[t];
    }
  }
}

// ---------------------------------------------------------------------------
// WMMA GEMM: C[M,N] = A[M,K] * BT[N,K]^T, f16 in / f32 accumulate.
// Block tile 128x256, K-step 32, 8 waves (2x4), wave tile 64x64 (4x4 WMMA).
// Double-buffered LDS fed by GLOBAL_LOAD_ASYNC_TO_LDS_B128 (ASYNCcnt).
// MODE 0: f32 to Cout.  MODE 1: fused QKV epilogue (scalers, q*=0.125,
//         V written transposed [B,H,hd,T]).
// ---------------------------------------------------------------------------
template <int MODE>
__global__ __launch_bounds__(256)
void gemm_wmma_kernel(const _Float16* __restrict__ A,
                      const _Float16* __restrict__ BT,
                      int M, int N, int K,
                      float* __restrict__ Cout,
                      _Float16* __restrict__ qh, _Float16* __restrict__ kh,
                      _Float16* __restrict__ vTh,
                      const float* __restrict__ qa, const float* __restrict__ qb2,
                      const float* __restrict__ ka, const float* __restrict__ kb2,
                      const float* __restrict__ va, const float* __restrict__ vb2,
                      int T, int H) {
  __shared__ _Float16 As[2][128 * 40];  // ld=40 halves: 16B-aligned rows, no conflicts
  __shared__ _Float16 Bs[2][256 * 40];
  const int tid  = threadIdx.x;
  const int bm   = blockIdx.y, bn = blockIdx.x;
  const int lane = tid & 31, wv = tid >> 5;
  const int n16  = lane & 15, g = lane >> 4;
  const int wm   = wv >> 2, wn = wv & 3;

  const v8f zero = {0.f, 0.f, 0.f, 0.f, 0.f, 0.f, 0.f, 0.f};
  v8f acc[4][4];
#pragma unroll
  for (int i = 0; i < 4; ++i)
#pragma unroll
    for (int j = 0; j < 4; ++j) acc[i][j] = zero;

  const _Float16* Ab = A + (size_t)(bm * 128) * K;
  const _Float16* Bb = BT + (size_t)(bn * 256) * K;

  // per-thread async-copy assignments: A 128x32 halves (2x16B), B 256x32 (4x16B)
  const int arow = tid >> 1, acol = (tid & 1) * 16;
  const _Float16* gA = Ab + (size_t)arow * K + acol;
  const _Float16* gB = Bb + (size_t)tid * K;
  unsigned lA[2], lB[2];
#pragma unroll
  for (int s = 0; s < 2; ++s) {
    lA[s] = (unsigned)(uintptr_t)&As[s][arow * 40 + acol];
    lB[s] = (unsigned)(uintptr_t)&Bs[s][tid * 40];
  }

#define ISSUE_STAGE(buf, k0)                                                   \
  do {                                                                         \
    async_ld16(lA[buf],      gA + (k0));                                       \
    async_ld16(lA[buf] + 16, gA + (k0) + 8);                                   \
    async_ld16(lB[buf],      gB + (k0));                                       \
    async_ld16(lB[buf] + 16, gB + (k0) + 8);                                   \
    async_ld16(lB[buf] + 32, gB + (k0) + 16);                                  \
    async_ld16(lB[buf] + 48, gB + (k0) + 24);                                  \
  } while (0)

  const int nIter = K >> 5;
  ISSUE_STAGE(0, 0);
  for (int it = 0; it < nIter; ++it) {
    const int cur = it & 1;
    if (it + 1 < nIter) {
      ISSUE_STAGE(cur ^ 1, (it + 1) << 5);
      asm volatile("s_wait_asynccnt 0x6" ::: "memory");  // stage `it` landed
    } else {
      asm volatile("s_wait_asynccnt 0x0" ::: "memory");
    }
    __syncthreads();

    AFragU bf[4];
#pragma unroll
    for (int j = 0; j < 4; ++j) {
      // B fragment: lane column n, halves i <-> K = 16g + i (contiguous)
      const _Float16* p = &Bs[cur][(wn * 64 + j * 16 + n16) * 40 + 16 * g];
      bf[j].u[0] = *(const uint4*)p;
      bf[j].u[1] = *(const uint4*)(p + 8);
    }
#pragma unroll
    for (int i = 0; i < 4; ++i) {
      // A fragment: lane row m, halves i <-> K = 16g + i (contiguous)
      AFragU af;
      const _Float16* p = &As[cur][(wm * 64 + i * 16 + n16) * 40 + 16 * g];
      af.u[0] = *(const uint4*)p;
      af.u[1] = *(const uint4*)(p + 8);
#pragma unroll
      for (int j = 0; j < 4; ++j) acc[i][j] = wmma_f16(af.v, bf[j].v, acc[i][j]);
    }
    __syncthreads();  // all waves done reading before next overwrite
  }
#undef ISSUE_STAGE

#pragma unroll
  for (int i = 0; i < 4; ++i) {
#pragma unroll
    for (int j = 0; j < 4; ++j) {
#pragma unroll
      for (int r = 0; r < 8; ++r) {
        // C/D layout: row m = r + 8g, col n = lane%16
        int grow = bm * 128 + wm * 64 + i * 16 + r + 8 * g;
        int gcol = bn * 256 + wn * 64 + j * 16 + n16;
        float val = acc[i][j][r];
        if (MODE == 0) {
          Cout[(size_t)grow * N + gcol] = val;
        } else {
          int bidx = grow / T, t = grow - bidx * T;
          int sec = gcol >> 10, cr = gcol & 1023;   // C = 1024
          int h = cr >> 6, d = cr & 63;             // hd = 64
          if (sec == 0) {
            float v2 = (fmaxf(qa[h], 1e-20f) * val + qb2[h]) * 0.125f; // fold hd^-0.5
            qh[((size_t)(bidx * H + h) * T + t) * 64 + d] = (_Float16)v2;
          } else if (sec == 1) {
            float v2 = fmaxf(ka[h], 1e-20f) * val + kb2[h];
            kh[((size_t)(bidx * H + h) * T + t) * 64 + d] = (_Float16)v2;
          } else {
            float v2 = fmaxf(va[h], 1e-20f) * val + vb2[h];
            vTh[((size_t)(bidx * H + h) * 64 + d) * T + t] = (_Float16)v2; // transposed
          }
        }
      }
    }
  }
}

// ---------------------------------------------------------------------------
// K3: sliding-window flash attention. One wave per (b, h, 16-query tile).
// Keys in blocks of 32: 4 WMMA for Q*K^T, online softmax in accumulator
// registers (shfl_xor row reductions, masking only on edge blocks), P bounced
// via LDS into A-fragment layout, 4 WMMA for P*V.
// ---------------------------------------------------------------------------
__global__ __launch_bounds__(128)
void attn_kernel(const _Float16* __restrict__ q, const _Float16* __restrict__ k,
                 const _Float16* __restrict__ vT, _Float16* __restrict__ yh,
                 const float* __restrict__ oa, const float* __restrict__ ob,
                 const int* __restrict__ winp, int B, int T, int H) {
  __shared__ _Float16 Pw[4][16 * 32];
  const int lane = threadIdx.x & 31, w = threadIdx.x >> 5;
  const int n16 = lane & 15, g = lane >> 4;
  const int qtiles = T >> 4;
  int wid = blockIdx.x * 4 + w;
  int qt = wid % qtiles;
  int bh = wid / qtiles;
  int h = bh % H, b = bh / H;
  const int q0 = qt * 16;
  const int wsz = winp[0];

  const _Float16* qbase = q + ((size_t)(b * H + h) * T) * 64;
  const _Float16* kbase = k + ((size_t)(b * H + h) * T) * 64;
  const _Float16* vbase = vT + ((size_t)(b * H + h) * 64) * T;

  AFragU aq[2];  // hd=64 -> two K=32 chunks
#pragma unroll
  for (int c = 0; c < 2; ++c) {
    const _Float16* p = qbase + (size_t)(q0 + n16) * 64 + c * 32 + 16 * g;
    aq[c].u[0] = *(const uint4*)p;
    aq[c].u[1] = *(const uint4*)(p + 8);
  }

  const v8f zero = {0.f, 0.f, 0.f, 0.f, 0.f, 0.f, 0.f, 0.f};
  v8f o_[4] = {zero, zero, zero, zero};
  float m_run[8], l_run[8];
#pragma unroll
  for (int r = 0; r < 8; ++r) { m_run[r] = -__builtin_inff(); l_run[r] = 0.0f; }

  int kb_lo = q0 - wsz + 1;
  if (kb_lo < 0) kb_lo = 0;
  kb_lo &= ~31;
  int kb_hi = (q0 + 15) & ~31;

  for (int kb0 = kb_lo; kb0 <= kb_hi; kb0 += 32) {
    if (kb0 + 32 <= kb_hi) {  // prefetch next key/value block
      __builtin_prefetch(kbase + (size_t)(kb0 + 32 + n16) * 64, 0, 0);
      __builtin_prefetch(vbase + (size_t)n16 * T + kb0 + 32, 0, 0);
    }
    v8f s0 = zero, s1 = zero;
#pragma unroll
    for (int c = 0; c < 2; ++c) {
      AFragU bk0, bk1;  // k stored [key][dim] == BT layout for Q*K^T
      const _Float16* p0 = kbase + (size_t)(kb0 + n16) * 64 + c * 32 + 16 * g;
      bk0.u[0] = *(const uint4*)p0;
      bk0.u[1] = *(const uint4*)(p0 + 8);
      const _Float16* p1 = kbase + (size_t)(kb0 + 16 + n16) * 64 + c * 32 + 16 * g;
      bk1.u[0] = *(const uint4*)p1;
      bk1.u[1] = *(const uint4*)(p1 + 8);
      s0 = wmma_f16(aq[c].v, bk0.v, s0);
      s1 = wmma_f16(aq[c].v, bk1.v, s1);
    }

    // interior blocks of the window need no mask (wave-uniform test)
    const bool full = (kb0 + 31 <= q0) && (kb0 >= q0 + 16 - wsz);

    float alpha[8];
#pragma unroll
    for (int r = 0; r < 8; ++r) {
      float e0, e1;
      if (full) {
        e0 = s0[r];
        e1 = s1[r];
      } else {
        int i = q0 + r + 8 * g;
        int j0 = kb0 + n16;
        int j1 = j0 + 16;
        e0 = ((j0 <= i) && (j0 > i - wsz)) ? s0[r] : -__builtin_inff();
        e1 = ((j1 <= i) && (j1 > i - wsz)) ? s1[r] : -__builtin_inff();
      }
      float tm = fmaxf(e0, e1);
      tm = fmaxf(tm, __shfl_xor(tm, 1, 32));
      tm = fmaxf(tm, __shfl_xor(tm, 2, 32));
      tm = fmaxf(tm, __shfl_xor(tm, 4, 32));
      tm = fmaxf(tm, __shfl_xor(tm, 8, 32));
      float mnew = fmaxf(m_run[r], tm);
      float msafe = fmaxf(mnew, -1e30f);  // fully-masked rows -> p = 0 (no NaN)
      float al = __expf(m_run[r] - msafe);
      float p0 = __expf(e0 - msafe);
      float p1 = __expf(e1 - msafe);
      float rs = p0 + p1;
      rs += __shfl_xor(rs, 1, 32);
      rs += __shfl_xor(rs, 2, 32);
      rs += __shfl_xor(rs, 4, 32);
      rs += __shfl_xor(rs, 8, 32);
      l_run[r] = l_run[r] * al + rs;
      m_run[r] = mnew;
      alpha[r] = al;
      // D-layout -> LDS (row m = r+8g, cols n16 / n16+16)
      Pw[w][(r + 8 * g) * 32 + n16] = (_Float16)p0;
      Pw[w][(r + 8 * g) * 32 + 16 + n16] = (_Float16)p1;
    }
#pragma unroll
    for (int r = 0; r < 8; ++r) {
      o_[0][r] *= alpha[r];
      o_[1][r] *= alpha[r];
      o_[2][r] *= alpha[r];
      o_[3][r] *= alpha[r];
    }
    asm volatile("s_wait_dscnt 0x0" ::: "memory");  // wave-internal LDS RAW

    AFragU ap;  // re-read P as A fragment (row = lane%16)
    const _Float16* pp = &Pw[w][n16 * 32 + 16 * g];
    ap.u[0] = *(const uint4*)pp;
    ap.u[1] = *(const uint4*)(pp + 8);

#pragma unroll
    for (int nt = 0; nt < 4; ++nt) {
      AFragU bv;  // vT stored [dim][key] == BT layout for P*V
      const _Float16* vp = vbase + (size_t)(nt * 16 + n16) * T + kb0 + 16 * g;
      bv.u[0] = *(const uint4*)vp;
      bv.u[1] = *(const uint4*)(vp + 8);
      o_[nt] = wmma_f16(ap.v, bv.v, o_[nt]);
    }
  }

  float oam = fmaxf(oa[h], 1e-20f), obm = ob[h];
#pragma unroll
  for (int r = 0; r < 8; ++r) {
    float inv = 1.0f / l_run[r];
    size_t rowb = (size_t)(b * T + q0 + r + 8 * g) * 1024 + (size_t)h * 64;
#pragma unroll
    for (int nt = 0; nt < 4; ++nt)
      yh[rowb + nt * 16 + n16] = (_Float16)(o_[nt][r] * inv * oam + obm);
  }
}

// ---------------------------------------------------------------------------
extern "C" void kernel_launch(void* const* d_in, const int* in_sizes, int n_in,
                              void* d_out, int out_size, void* d_ws, size_t ws_size,
                              hipStream_t stream) {
  const float* x      = (const float*)d_in[0];
  const float* w_attn = (const float*)d_in[1];
  const float* w_proj = (const float*)d_in[2];
  const float* qa = (const float*)d_in[3];
  const float* qb = (const float*)d_in[4];
  const float* ka = (const float*)d_in[5];
  const float* kb = (const float*)d_in[6];
  const float* va = (const float*)d_in[7];
  const float* vb = (const float*)d_in[8];
  const float* oa = (const float*)d_in[9];
  const float* ob = (const float*)d_in[10];
  const int* win  = (const int*)d_in[11];

  const int Bb = 2, T = 2048, C = 1024, H = 16;
  const int M = Bb * T;    // 4096
  const int N3 = 3 * C;    // 3072

  _Float16* ws = (_Float16*)d_ws;
  size_t off = 0;
  _Float16* xh   = ws + off; off += (size_t)M * C;
  _Float16* wahT = ws + off; off += (size_t)N3 * C;
  _Float16* wpjT = ws + off; off += (size_t)C * C;
  _Float16* qh   = ws + off; off += (size_t)M * C;
  _Float16* kh   = ws + off; off += (size_t)M * C;
  _Float16* vTh  = ws + off; off += (size_t)M * C;
  _Float16* yh   = ws + off;

  // K1: casts + weight transposes
  int total = M * C + N3 * C + C * C;
  cast_pack_kernel<<<(total + 255) / 256, 256, 0, stream>>>(
      x, w_attn, w_proj, xh, wahT, wpjT, M * C, N3 * C, C * C, C, N3);

  // K2: QKV GEMM with fused scaling epilogue (async double-buffered)
  dim3 g2(N3 / 256, M / 128);
  gemm_wmma_kernel<1><<<g2, 256, 0, stream>>>(
      xh, wahT, M, N3, C, nullptr, qh, kh, vTh, qa, qb, ka, kb, va, vb, T, H);

  // K3: sliding-window flash attention (one wave per 16-query tile)
  attn_kernel<<<(Bb * H * (T / 16)) / 4, 128, 0, stream>>>(
      qh, kh, vTh, yh, oa, ob, win, Bb, T, H);

  // K4: output projection -> fp32 d_out
  dim3 g4(C / 256, M / 128);
  gemm_wmma_kernel<0><<<g4, 256, 0, stream>>>(
      yh, wpjT, M, C, C, (float*)d_out, nullptr, nullptr, nullptr,
      nullptr, nullptr, nullptr, nullptr, nullptr, nullptr, T, H);
}
